// HG_GNN_54150947668721
// MI455X (gfx1250) — matrix operand order repaired and checked by reading, not compiled, not run on hardware
//
#include <hip/hip_runtime.h>

#define EM        128
#define NNODES    120000
#define NEDGES    1920000
#define BSZ       1024
#define SL        10
#define NITEMS    119999

typedef __attribute__((ext_vector_type(16))) __bf16 v16bf;
typedef __attribute__((ext_vector_type(8)))  float  v8f;

__device__ __forceinline__ unsigned short f2bf(float x) {
  union { float f; unsigned int u; } a; a.f = x;
  unsigned int r = a.u + 0x7FFFu + ((a.u >> 16) & 1u);   // round-to-nearest-even
  return (unsigned short)(r >> 16);
}

__device__ __forceinline__ v8f wmma_bf16(v16bf a, v16bf b, v8f c) {
  return __builtin_amdgcn_wmma_f32_16x16x32_bf16(false, a, false, b, (short)0, c, false, false);
}

// ---------------- zero workspace (agg + deg) ----------------
__global__ void zero_kernel(float4* p, int n4) {
  int i = blockIdx.x * 256 + threadIdx.x;
  if (i < n4) p[i] = make_float4(0.f, 0.f, 0.f, 0.f);
}

// ---------------- edge scatter-add (mean aggregation numerator + degree) ----------------
__global__ void edge_kernel(const int* __restrict__ src, const int* __restrict__ dst,
                            const float* __restrict__ v2e, float* agg, float* deg) {
  int tid = blockIdx.x * 256 + threadIdx.x;      // NEDGES*32 = 61,440,000 exact
  int e = tid >> 5, c = tid & 31;
  if (e >= NEDGES) return;
  int s = src[e], d = dst[e];
  float4 x = ((const float4*)(v2e + (size_t)s * EM))[c];
  float* a = agg + (size_t)d * EM + c * 4;
  atomicAdd(a + 0, x.x); atomicAdd(a + 1, x.y);
  atomicAdd(a + 2, x.z); atomicAdd(a + 3, x.w);
  if (c == 0) atomicAdd(deg + d, 1.0f);
}

// ---------------- pack W_self / W_neigh into WMMA B-fragment order (bf16) ----------------
// B frag order packed here (16x16x32 bf16, contiguous k, matches the A loads): VGPR j, lanes0-15 -> K=2j,2j+1 ; lanes16-31 -> K=16+2j,17+2j
__global__ void wfrag_kernel(const float* __restrict__ Ws, const float* __restrict__ Wn,
                             unsigned short* wfrag) {
  int tid = blockIdx.x * 256 + threadIdx.x;      // 2048 = mat(2) x ks(4) x nt(8) x lane(32)
  if (tid >= 2048) return;
  int lane = tid & 31, nt = (tid >> 5) & 7, ks = (tid >> 8) & 3, mat = tid >> 10;
  const float* W = mat ? Wn : Ws;
  int col = nt * 16 + (lane & 15);
  int hi = (lane >> 4) & 1;
  unsigned short* dstp = wfrag + (size_t)(((mat * 4 + ks) * 8 + nt) * 32 + lane) * 16;
#pragma unroll
  for (int j = 0; j < 8; ++j) {
    int k0 = ks * 32 + hi * 16 + 2 * j;
    dstp[2 * j]     = f2bf(W[k0 * EM + col]);
    dstp[2 * j + 1] = f2bf(W[(k0 + 1) * EM + col]);
  }
}

// ---------------- v2e -> bf16 (B operand for scores GEMM) ----------------
__global__ void cvt_kernel(const float* __restrict__ v2e, unsigned short* v2eb) {
  int i = blockIdx.x * 256 + threadIdx.x;        // 3,840,000 exact (float4 units)
  if (i >= NNODES * EM / 4) return;
  float4 x = ((const float4*)v2e)[i];
  unsigned int lo = (unsigned int)f2bf(x.x) | ((unsigned int)f2bf(x.y) << 16);
  unsigned int hi = (unsigned int)f2bf(x.z) | ((unsigned int)f2bf(x.w) << 16);
  ((uint2*)v2eb)[i] = make_uint2(lo, hi);
}

// A frag (16x32 bf16): elems0-15 = K hi*16+0..15
__device__ __forceinline__ v16bf load_afrag_f32(const float* p, int hi, float scale) {
  union { v16bf v; unsigned short u[16]; } r;
  float4 a0 = *(const float4*)(p + hi * 16);
  float4 a1 = *(const float4*)(p + hi * 16 + 4);
  float4 b0 = *(const float4*)(p + hi * 16 + 8);
  float4 b1 = *(const float4*)(p + hi * 16 + 12);
  r.u[0]=f2bf(a0.x*scale); r.u[1]=f2bf(a0.y*scale); r.u[2]=f2bf(a0.z*scale); r.u[3]=f2bf(a0.w*scale);
  r.u[4]=f2bf(a1.x*scale); r.u[5]=f2bf(a1.y*scale); r.u[6]=f2bf(a1.z*scale); r.u[7]=f2bf(a1.w*scale);
  r.u[8]=f2bf(b0.x*scale); r.u[9]=f2bf(b0.y*scale); r.u[10]=f2bf(b0.z*scale); r.u[11]=f2bf(b0.w*scale);
  r.u[12]=f2bf(b1.x*scale); r.u[13]=f2bf(b1.y*scale); r.u[14]=f2bf(b1.z*scale); r.u[15]=f2bf(b1.w*scale);
  return r.v;
}

// ---------------- h1 = relu(v2e@Ws + (agg/deg)@Wn + b), written in-place over agg ----------------
__global__ void __launch_bounds__(128) h1_kernel(const float* __restrict__ v2e, float* agg,
                                                 const float* __restrict__ deg,
                                                 const unsigned short* __restrict__ wfrag,
                                                 const float* __restrict__ sage_b) {
  int wave = threadIdx.x >> 5, lane = threadIdx.x & 31;
  int rt = blockIdx.x * 4 + wave;                // 7500 row tiles
  int m = lane & 15, hi = lane >> 4;
  int row = rt * 16 + m;
  float rdeg = 1.0f / fmaxf(deg[row], 1.0f);
  const float* vrow = v2e + (size_t)row * EM;
  float* arow = agg + (size_t)row * EM;
  const v16bf* wf = (const v16bf*)wfrag;

  v8f acc[8];
  v8f z = {0.f,0.f,0.f,0.f,0.f,0.f,0.f,0.f};
#pragma unroll
  for (int i = 0; i < 8; ++i) acc[i] = z;

#pragma unroll
  for (int ks = 0; ks < 4; ++ks) {
    v16bf aS = load_afrag_f32(vrow + ks * 32, hi, 1.0f);
    v16bf aN = load_afrag_f32(arow + ks * 32, hi, rdeg);
#pragma unroll
    for (int nt = 0; nt < 8; ++nt) {
      v16bf bS = wf[((0 * 4 + ks) * 8 + nt) * 32 + lane];
      v16bf bN = wf[((1 * 4 + ks) * 8 + nt) * 32 + lane];
      acc[nt] = wmma_bf16(aS, bS, acc[nt]);
      acc[nt] = wmma_bf16(aN, bN, acc[nt]);
    }
  }
  // epilogue: C layout: VGPR r -> row r (lanes0-15) / r+8 (lanes16-31), col = lane&15
#pragma unroll
  for (int nt = 0; nt < 8; ++nt) {
    int col = nt * 16 + m;
    float bias = sage_b[col];
#pragma unroll
    for (int r = 0; r < 8; ++r) {
      int orow = rt * 16 + r + 8 * hi;
      agg[(size_t)orow * EM + col] = fmaxf(acc[nt][r] + bias, 0.0f);
    }
  }
}

// ---------------- gather node/user embeds, hs mean, hsg = hs@glu2_w, ueg = ue@glu4_w ----------------
__global__ void gather_kernel(const int* __restrict__ seq, const int* __restrict__ user,
                              const int* __restrict__ item_num, const float* __restrict__ mask,
                              const float* __restrict__ h1, const float* __restrict__ v2e,
                              const float* __restrict__ glu2_w, const float* __restrict__ glu4_w,
                              float* node, float* usere, float* hsg, float* ueg) {
  int b = blockIdx.x, t = threadIdx.x;
  __shared__ float sh[EM];
  float acc = 0.f, msum = 0.f;
#pragma unroll
  for (int l = 0; l < SL; ++l) {
    int idx = seq[b * SL + l];
    float nv = 0.5f * (h1[(size_t)idx * EM + t] + v2e[(size_t)idx * EM + t]);
    node[(size_t)(b * SL + l) * EM + t] = nv;
    float mk = mask[b * SL + l];
    acc += nv * mk; msum += mk;
  }
  float hs = acc / msum;
  int uidx = user[b] + item_num[0];
  float ue = 0.5f * (h1[(size_t)uidx * EM + t] + v2e[(size_t)uidx * EM + t]);
  usere[b * EM + t] = ue;
  sh[t] = hs; __syncthreads();
  float s1 = 0.f;
#pragma unroll 8
  for (int k = 0; k < EM; ++k) s1 += sh[k] * glu2_w[k * EM + t];
  hsg[b * EM + t] = s1;
  __syncthreads(); sh[t] = ue; __syncthreads();
  float s2 = 0.f;
#pragma unroll 8
  for (int k = 0; k < EM; ++k) s2 += sh[k] * glu4_w[k * EM + t];
  ueg[b * EM + t] = s2;
}

// ---------------- attention: beta/beta2 scalars per (b,l) ----------------
__global__ void att_kernel(const float* __restrict__ node, const float* __restrict__ pos_emb,
                           const int* __restrict__ pos_idx, const float* __restrict__ w1,
                           const float* __restrict__ glu1_w, const float* __restrict__ glu1_b,
                           const float* __restrict__ hsg, const float* __restrict__ w2,
                           const float* __restrict__ w3, const float* __restrict__ glu3_w,
                           const float* __restrict__ glu3_b, const float* __restrict__ ueg,
                           const float* __restrict__ w4, const float* __restrict__ mask,
                           float* beta, float* beta2) {
  int bl = blockIdx.x, b = bl / SL, t = threadIdx.x;
  __shared__ float xc[2 * EM];
  __shared__ float nh[EM];
  __shared__ float red[4];
  xc[t] = pos_emb[(size_t)pos_idx[bl] * EM + t];
  xc[EM + t] = node[(size_t)bl * EM + t];
  __syncthreads();
  float a1 = 0.f;
#pragma unroll 8
  for (int k = 0; k < 2 * EM; ++k) a1 += xc[k] * w1[k * EM + t];
  a1 = tanhf(a1);
  nh[t] = a1; __syncthreads();
  float g = glu1_b[t] + hsg[b * EM + t];
#pragma unroll 8
  for (int k = 0; k < EM; ++k) g += nh[k] * glu1_w[k * EM + t];
  float s = 1.f / (1.f + expf(-g));
  float c1 = s * w2[t];
#pragma unroll
  for (int off = 16; off; off >>= 1) c1 += __shfl_down(c1, off, 32);
  if ((t & 31) == 0) red[t >> 5] = c1;
  __syncthreads();
  if (t == 0) beta[bl] = (red[0] + red[1] + red[2] + red[3]) * mask[bl];
  __syncthreads();
  float a2 = 0.f;
#pragma unroll 8
  for (int k = 0; k < EM; ++k) a2 += xc[EM + k] * w3[k * EM + t];
  a2 = tanhf(a2);
  nh[t] = a2; __syncthreads();
  float g2 = glu3_b[t] + ueg[b * EM + t];
#pragma unroll 8
  for (int k = 0; k < EM; ++k) g2 += nh[k] * glu3_w[k * EM + t];
  float s2 = 1.f / (1.f + expf(-g2));
  float c2 = s2 * w4[t];
#pragma unroll
  for (int off = 16; off; off >>= 1) c2 += __shfl_down(c2, off, 32);
  if ((t & 31) == 0) red[t >> 5] = c2;
  __syncthreads();
  if (t == 0) beta2[bl] = (red[0] + red[1] + red[2] + red[3]) * mask[bl];
}

// ---------------- session fuse -> seq_embeds (bf16 A for scores GEMM) ----------------
__global__ void sess_kernel(const float* __restrict__ node, const float* __restrict__ usere,
                            const float* __restrict__ beta, const float* __restrict__ beta2,
                            const float* __restrict__ sc_w, const float* __restrict__ sc_b,
                            unsigned short* abf) {
  int b = blockIdx.x, t = threadIdx.x;
  __shared__ float red[4];
  __shared__ float sh_alpha;
  float sv = 0.f, su = 0.f;
#pragma unroll
  for (int l = 0; l < SL; ++l) {
    float nd = node[(size_t)(b * SL + l) * EM + t];
    sv += beta[b * SL + l] * nd;
    su += beta2[b * SL + l] * nd;
  }
  float c = sv * sc_w[t] + su * sc_w[EM + t];
#pragma unroll
  for (int off = 16; off; off >>= 1) c += __shfl_down(c, off, 32);
  if ((t & 31) == 0) red[t >> 5] = c;
  __syncthreads();
  if (t == 0) sh_alpha = 1.f / (1.f + expf(-(red[0] + red[1] + red[2] + red[3] + sc_b[0])));
  __syncthreads();
  float al = sh_alpha;
  float se = usere[b * EM + t] + al * sv + (1.f - al) * su;
  abf[b * EM + t] = f2bf(se);
}

// ---------------- scores = seq_embeds @ v2e[1:].T  (1024 x 119999, K=128) ----------------
__global__ void __launch_bounds__(256) scores_kernel(const unsigned short* __restrict__ abf,
                                                     const unsigned short* __restrict__ v2eb,
                                                     float* __restrict__ out) {
  int wave = threadIdx.x >> 5, lane = threadIdx.x & 31;
  int m = lane & 15, hi = lane >> 4;
  int colbase = blockIdx.x * 32;                 // 3750 blocks x 32 cols

  v8f acc[16];
  v8f z = {0.f,0.f,0.f,0.f,0.f,0.f,0.f,0.f};
#pragma unroll
  for (int i = 0; i < 16; ++i) acc[i] = z;

#pragma unroll
  for (int ks = 0; ks < 4; ++ks) {
    v16bf bfr[2];
#pragma unroll
    for (int c = 0; c < 2; ++c) {
      int n = colbase + c * 16 + m;
      int vr = (n + 1 < NNODES) ? (n + 1) : (NNODES - 1);
      const uint4* p = (const uint4*)(v2eb + (size_t)vr * EM + ks * 32 + hi * 16);
      union { v16bf v; uint4 q[2]; } u;
      u.q[0] = p[0]; u.q[1] = p[1];
      bfr[c] = u.v;
    }
#pragma unroll
    for (int i = 0; i < 8; ++i) {
      int arow = (wave * 8 + i) * 16 + m;
      const unsigned short* pa = abf + (size_t)arow * EM + ks * 32;
      union { v16bf v; uint4 q[2]; } u;
      u.q[0] = *(const uint4*)(pa + hi * 16);
      u.q[1] = *(const uint4*)(pa + hi * 16 + 8);
      acc[i * 2 + 0] = wmma_bf16(u.v, bfr[0], acc[i * 2 + 0]);
      acc[i * 2 + 1] = wmma_bf16(u.v, bfr[1], acc[i * 2 + 1]);
    }
  }
#pragma unroll
  for (int i = 0; i < 8; ++i) {
    int rbase = (wave * 8 + i) * 16;
#pragma unroll
    for (int c = 0; c < 2; ++c) {
      int n = colbase + c * 16 + m;
      if (n < NITEMS) {
#pragma unroll
        for (int r = 0; r < 8; ++r) {
          int row = rbase + r + 8 * hi;
          out[(size_t)row * NITEMS + n] = acc[i * 2 + c][r];
        }
      }
    }
  }
}

extern "C" void kernel_launch(void* const* d_in, const int* in_sizes, int n_in,
                              void* d_out, int out_size, void* d_ws, size_t ws_size,
                              hipStream_t stream) {
  const int*   src      = (const int*)d_in[0];
  const int*   dst      = (const int*)d_in[1];
  const int*   user     = (const int*)d_in[2];
  const int*   seq      = (const int*)d_in[3];
  const float* mask     = (const float*)d_in[4];
  /* d_in[5] seq_len unused by reference math */
  const int*   pos_idx  = (const int*)d_in[6];
  const int*   item_num = (const int*)d_in[7];
  const float* v2e      = (const float*)d_in[8];
  const float* W_self   = (const float*)d_in[9];
  const float* W_neigh  = (const float*)d_in[10];
  const float* sage_b   = (const float*)d_in[11];
  const float* pos_emb  = (const float*)d_in[12];
  const float* w1       = (const float*)d_in[13];
  const float* w2       = (const float*)d_in[14];
  const float* glu1_w   = (const float*)d_in[15];
  const float* glu1_b   = (const float*)d_in[16];
  const float* glu2_w   = (const float*)d_in[17];
  const float* w3       = (const float*)d_in[18];
  const float* w4       = (const float*)d_in[19];
  const float* glu3_w   = (const float*)d_in[20];
  const float* glu3_b   = (const float*)d_in[21];
  const float* glu4_w   = (const float*)d_in[22];
  const float* sc_w     = (const float*)d_in[23];
  const float* sc_b     = (const float*)d_in[24];
  float* out = (float*)d_out;

  // workspace layout (all offsets 256B aligned), ~99.9 MB total
  char* ws = (char*)d_ws;
  float*          agg   = (float*)(ws + 0);           // 61,440,000 B (agg -> h1 in-place)
  float*          deg   = (float*)(ws + 61440000);    //    480,000 B
  unsigned short* v2eb  = (unsigned short*)(ws + 61920000);  // 30,720,000 B
  unsigned short* wfrag = (unsigned short*)(ws + 92640000);  //     65,536 B
  float*          node  = (float*)(ws + 92705536);    //  5,242,880 B
  float*          usere = (float*)(ws + 97948416);    //    524,288 B
  float*          hsg   = (float*)(ws + 98472704);    //    524,288 B
  float*          ueg   = (float*)(ws + 98996992);    //    524,288 B
  float*          beta  = (float*)(ws + 99521280);    //     40,960 B
  float*          beta2 = (float*)(ws + 99562240);    //     40,960 B
  unsigned short* abf   = (unsigned short*)(ws + 99603200); // 262,144 B

  zero_kernel <<<15118, 256, 0, stream>>>((float4*)agg, 3870000); // agg + deg
  edge_kernel <<<240000, 256, 0, stream>>>(src, dst, v2e, agg, deg);
  wfrag_kernel<<<8, 256, 0, stream>>>(W_self, W_neigh, wfrag);
  cvt_kernel  <<<15000, 256, 0, stream>>>(v2e, v2eb);
  h1_kernel   <<<1875, 128, 0, stream>>>(v2e, agg, deg, wfrag, sage_b);
  gather_kernel<<<BSZ, 128, 0, stream>>>(seq, user, item_num, mask, agg, v2e,
                                         glu2_w, glu4_w, node, usere, hsg, ueg);
  att_kernel  <<<BSZ * SL, 128, 0, stream>>>(node, pos_emb, pos_idx, w1, glu1_w, glu1_b,
                                             hsg, w2, w3, glu3_w, glu3_b, ueg, w4, mask,
                                             beta, beta2);
  sess_kernel <<<BSZ, 128, 0, stream>>>(node, usere, beta, beta2, sc_w, sc_b, abf);
  scores_kernel<<<3750, 256, 0, stream>>>(abf, v2eb, out);
}
